// TorchMLAMiniCPMAttention_83820581749276
// MI455X (gfx1250) — matrix operand order repaired and checked by reading, not compiled, not run on hardware
//
#include <hip/hip_runtime.h>
#include <hip/hip_bf16.h>

// ---------------------------------------------------------------------------
// MLA (MiniCPM / DeepSeek-style absorbed) attention for gfx1250, bf16 WMMA,
// async global->LDS staging, double-buffered GEMM tiles.
// B=2, S=2048, HID=2048, H=16, Q_LORA=1536, KV_LORA=512, ROPE=64, VDIM=128
// ---------------------------------------------------------------------------

typedef unsigned short u16;
typedef unsigned int   u32;
typedef __attribute__((ext_vector_type(16))) __bf16 v16bf;
typedef __attribute__((ext_vector_type(8)))  float  v8f;

#define B_    2
#define S_    2048
#define HID_  2048
#define H_    16
#define QLORA 1536
#define KVL   512
#define ROPE_ 64
#define VDIM_ 128
#define SCALE_QK 0.07216878364870323f  /* 1/sqrt(128+64) */

// ---------------- async global->LDS (CDNA5) ----------------

#if defined(__has_builtin)
#if __has_builtin(__builtin_amdgcn_global_load_async_to_lds_b128) && \
    __has_builtin(__builtin_amdgcn_s_wait_asynccnt)
#define MLA_ASYNC 1
#endif
#endif
#ifndef MLA_ASYNC
#define MLA_ASYNC 0
#endif

typedef int v4i __attribute__((vector_size(16)));
typedef __attribute__((address_space(1))) v4i gv4i;
typedef __attribute__((address_space(3))) v4i lv4i;

static __device__ __forceinline__ void cp_async16(void* lds, const void* g) {
#if MLA_ASYNC
  __builtin_amdgcn_global_load_async_to_lds_b128((gv4i*)g, (lv4i*)lds, 0, 0);
#else
  *(uint4*)lds = *(const uint4*)g;
#endif
}
static __device__ __forceinline__ void async_wait0() {
#if MLA_ASYNC
  __builtin_amdgcn_s_wait_asynccnt(0);
#endif
}
static __device__ __forceinline__ void async_wait3() {
#if MLA_ASYNC
  __builtin_amdgcn_s_wait_asynccnt(3);
#endif
}

// ---------------- helpers ----------------

static __device__ __forceinline__ u16 f2bf(float f) {
  u32 u = __builtin_bit_cast(u32, f);
  u += 0x7FFFu + ((u >> 16) & 1u);
  return (u16)(u >> 16);
}

struct U32x8 { uint4 a, b; };

static __device__ __forceinline__ v16bf bc16(U32x8 v) {
  return __builtin_bit_cast(v16bf, v);
}

// A-operand fragment (16x32 bf16): lane L<16 -> row L, K in {k0..k0+7, k0+16..k0+23}
//                                  lane L>=16 -> row L-16, K in {k0+8..15, k0+24..31}
static __device__ __forceinline__ v16bf frag_a_ld(const u16* base, int ld, int row0, int k0) {
  int lane = threadIdx.x & 31;
  const u16* p = base + (row0 + (lane & 15)) * ld + k0 + ((lane >> 4) << 3);
  U32x8 r;
  r.a = *(const uint4*)p;
  r.b = *(const uint4*)(p + 16);
  return bc16(r);
}

// B-operand fragment (32x16 bf16, stored as [N][K] rows): lane L<16 -> col L, K in {k0..k0+7, k0+16..k0+23}
//                                                         lane L>=16 -> col L-16, K in {k0+8..15, k0+24..31}
static __device__ __forceinline__ v16bf frag_b_ld(const u16* base, int ld, int n0, int k0) {
  int lane = threadIdx.x & 31;
  const u16* p = base + (n0 + (lane & 15)) * ld + k0 + ((lane >> 4) << 3);
  U32x8 r;
  r.a = *(const uint4*)p;
  r.b = *(const uint4*)(p + 16);
  return bc16(r);
}

static __device__ __forceinline__ v8f wmma_bf16(v16bf a, v16bf b, v8f c) {
  return __builtin_amdgcn_wmma_f32_16x16x32_bf16(false, a, false, b, (short)0, c,
                                                 false, false);
}

// ---------------- elementwise conversion ----------------

__global__ __launch_bounds__(256) void mla_f2bf_kernel(const float* __restrict__ in,
                                                       u16* __restrict__ out, long n) {
  long i = (long)blockIdx.x * 256 + threadIdx.x;
  long stride = (long)gridDim.x * 256;
  for (; i < n; i += stride) out[i] = f2bf(in[i]);
}

// ------- generic 128x64 WMMA GEMM, double-buffered LDS: C = A[M,K] * W[N,K]^T -------

#define MODE_F32   0
#define MODE_BF16  1
#define MODE_NOPE  2   /* out[( (row/S)*H + col/512 )*S + row%S ][ col%512 ]  (bf16) */
#define MODE_VPROJ 3   /* z=b*H+h; out[b][row][h*128+col] (bf16, row-stride 2048)   */

__global__ __launch_bounds__(256)
void mla_gemm_kernel(const u16* __restrict__ A, const u16* __restrict__ W,
                     float* __restrict__ Cf, u16* __restrict__ Cb,
                     int M, int N, int K, int lda, int ldw, int ldc,
                     long aBatch, long wBatch, long cBatch, int mode) {
  __shared__ u16 sA[2][128 * 40];
  __shared__ u16 sB[2][64 * 40];

  int z = blockIdx.z;
  int vb = 0, vh = 0;
  if (mode == MODE_VPROJ) {
    vb = z / H_;
    vh = z % H_;
    A += (long)z * aBatch;
    W += (long)vh * wBatch;
  } else {
    A += (long)z * aBatch;
    W += (long)z * wBatch;
  }

  int m0 = blockIdx.y * 128;
  int n0 = blockIdx.x * 64;
  int tid = threadIdx.x;
  int wv = tid >> 5, lane = tid & 31;
  int wr = wv & 3, wc = wv >> 2;
  int half = lane >> 4, lr = lane & 15;

  // per-thread: 2 A-chunks + 1 B-chunk = 3 async copies per k-tile
  auto copy_tile = [&](int buf, int kt) {
#pragma unroll
    for (int i = tid; i < 512; i += 256) {           // A: 128 rows x 4 segs
      int r = i >> 2, sg = i & 3;
      cp_async16(&sA[buf][r * 40 + sg * 8],
                 A + (long)(m0 + r) * lda + kt + sg * 8);
    }
    {                                                // B: 64 rows x 4 segs
      int r = tid >> 2, sg = tid & 3;
      cp_async16(&sB[buf][r * 40 + sg * 8],
                 W + (long)(n0 + r) * ldw + kt + sg * 8);
    }
  };

  v8f zero = {0.f, 0.f, 0.f, 0.f, 0.f, 0.f, 0.f, 0.f};
  v8f acc[4];
#pragma unroll
  for (int j = 0; j < 4; j++) acc[j] = zero;

  int nk = K >> 5;
  copy_tile(0, 0);
  for (int t = 0; t < nk; t++) {
    int cur = t & 1;
    if (t + 1 < nk) {
      copy_tile(cur ^ 1, (t + 1) << 5);
      async_wait3();
    } else {
      async_wait0();
    }
    __syncthreads();
    v16bf a0 = frag_a_ld(sA[cur], 40, wr * 32, 0);
    v16bf a1 = frag_a_ld(sA[cur], 40, wr * 32 + 16, 0);
    v16bf b0 = frag_b_ld(sB[cur], 40, wc * 32, 0);
    v16bf b1 = frag_b_ld(sB[cur], 40, wc * 32 + 16, 0);
    acc[0] = wmma_bf16(a0, b0, acc[0]);
    acc[1] = wmma_bf16(a0, b1, acc[1]);
    acc[2] = wmma_bf16(a1, b0, acc[2]);
    acc[3] = wmma_bf16(a1, b1, acc[3]);
    __syncthreads();
  }

#pragma unroll
  for (int at = 0; at < 2; at++) {
#pragma unroll
    for (int s2 = 0; s2 < 2; s2++) {
      v8f a = acc[at * 2 + s2];
      int col = n0 + wc * 32 + s2 * 16 + lr;
#pragma unroll
      for (int i = 0; i < 8; i++) {
        int row = m0 + wr * 32 + at * 16 + i + half * 8;
        float v = a[i];
        if (mode == MODE_F32) {
          Cf[(long)z * cBatch + (long)row * ldc + col] = v;
        } else if (mode == MODE_BF16) {
          Cb[(long)z * cBatch + (long)row * ldc + col] = f2bf(v);
        } else if (mode == MODE_NOPE) {
          int bb = row >> 11, ss = row & (S_ - 1);
          int hh = col >> 9, dd = col & (KVL - 1);
          Cb[(((long)(bb * H_ + hh) * S_ + ss) << 9) + dd] = f2bf(v);
        } else {  // MODE_VPROJ
          Cb[(long)vb * S_ * (H_ * VDIM_) + (long)row * (H_ * VDIM_) +
             vh * VDIM_ + col] = f2bf(v);
        }
      }
    }
  }
}

// ---------------- RMSNorm (q_a path): f32 in -> bf16 out ----------------

__global__ __launch_bounds__(256)
void mla_rmsnorm_kernel(const float* __restrict__ in, const float* __restrict__ w,
                        u16* __restrict__ out, int n) {
  __shared__ float red[256];
  int row = blockIdx.x, tid = threadIdx.x;
  const float* x = in + (long)row * n;
  float ss = 0.f;
  for (int i = tid; i < n; i += 256) ss += x[i] * x[i];
  red[tid] = ss;
  __syncthreads();
  for (int s = 128; s > 0; s >>= 1) {
    if (tid < s) red[tid] += red[tid + s];
    __syncthreads();
  }
  float sc = rsqrtf(red[0] / (float)n + 1e-6f);
  for (int i = tid; i < n; i += 256) out[(long)row * n + i] = f2bf(x[i] * sc * w[i]);
}

// ---------------- kv post: rmsnorm(ckv) + rope(k_rope) ----------------

__global__ __launch_bounds__(256)
void mla_kvpost_kernel(const float* __restrict__ ckvf, const float* __restrict__ lnw,
                       u16* __restrict__ ckv_b, u16* __restrict__ kr_b) {
  __shared__ float red[256];
  int row = blockIdx.x, tid = threadIdx.x;
  const float* x = ckvf + (long)row * (KVL + ROPE_);
  float ss = 0.f;
  for (int i = tid; i < KVL; i += 256) ss += x[i] * x[i];
  red[tid] = ss;
  __syncthreads();
  for (int s = 128; s > 0; s >>= 1) {
    if (tid < s) red[tid] += red[tid + s];
    __syncthreads();
  }
  float sc = rsqrtf(red[0] / (float)KVL + 1e-6f);
  for (int i = tid; i < KVL; i += 256)
    ckv_b[(long)row * KVL + i] = f2bf(x[i] * sc * lnw[i]);
  if (tid < ROPE_) {
    int d = tid;
    int pos = row & (S_ - 1);
    float inv = powf(10000.f, -(float)(d & 31) / 32.f);
    float ang = (float)pos * inv;
    float c = cosf(ang), s = sinf(ang);
    float v = x[KVL + d];
    float rot = (d < 32) ? -x[KVL + d + 32] : x[KVL + d - 32];
    kr_b[(long)row * ROPE_ + d] = f2bf(v * c + rot * s);
  }
}

// ---------------- q_rope post: rope + [B,H,S,64] transpose ----------------

__global__ __launch_bounds__(256)
void mla_qrope_kernel(const float* __restrict__ qrf, u16* __restrict__ out) {
  long idx = (long)blockIdx.x * 256 + threadIdx.x;
  if (idx >= (long)(B_ * S_) * (H_ * ROPE_)) return;
  int row = (int)(idx >> 10);          // token
  int c   = (int)(idx & 1023);
  int h = c >> 6, d = c & 63;
  int b = row >> 11, s = row & (S_ - 1);
  float inv = powf(10000.f, -(float)(d & 31) / 32.f);
  float ang = (float)s * inv;
  float cs = cosf(ang), sn = sinf(ang);
  const float* rowp = qrf + (long)row * (H_ * ROPE_) + h * ROPE_;
  float v = rowp[d];
  float rot = (d < 32) ? -rowp[d + 32] : rowp[d - 32];
  out[(((long)(b * H_ + h) * S_ + s) << 6) + d] = f2bf(v * cs + rot * sn);
}

// ---------------- flash attention over compressed KV ----------------
// grid: (S/64, H, B). block 256 (8 waves). Dynamic LDS ~249KB.

#define SQ_LD 520
#define SR_LD 72
#define SS_LD 68

__global__ __launch_bounds__(256)
void mla_attn_kernel(const u16* __restrict__ nq, const u16* __restrict__ qr,
                     const u16* __restrict__ ckv, const u16* __restrict__ kr,
                     u16* __restrict__ ctx) {
  extern __shared__ char smem[];
  u16* sQ  = (u16*)smem;                 // 64 x SQ_LD
  u16* sQr = sQ + 64 * SQ_LD;            // 64 x SR_LD
  u16* sK  = sQr + 64 * SR_LD;           // 64 x SQ_LD   (keys x feat)
  u16* sKr = sK + 64 * SQ_LD;            // 64 x SR_LD
  u16* sKT = sKr + 64 * SR_LD;           // 512 x SR_LD  (feat x keys)
  u16* sP  = sKT + 512 * SR_LD;          // 64 x SR_LD
  float* sS   = (float*)(sP + 64 * SR_LD);  // 64 x SS_LD
  float* sM   = sS + 64 * SS_LD;
  float* sL   = sM + 64;
  float* sAl  = sL + 64;
  float* sRed = sAl + 64;                // 64 x 4
  float* sSum = sRed + 256;              // 64 x 4

  int qt = blockIdx.x, hh = blockIdx.y, bb = blockIdx.z;
  long qbase = (long)(bb * H_ + hh) * S_ + qt * 64;
  const u16* nqp = nq + qbase * KVL;
  const u16* qrp = qr + qbase * ROPE_;
  const u16* ckp = ckv + (long)bb * S_ * KVL;
  const u16* krp = kr + (long)bb * S_ * ROPE_;

  int tid = threadIdx.x;
  int wv = tid >> 5, lane = tid & 31;
  int wr = wv & 3, wc = wv >> 2;
  int half = lane >> 4, lr = lane & 15;

  auto issue_k = [&](int kt) {
    int k0 = kt * 64;
    for (int i = tid; i < 64 * 64; i += 256) {
      int r = i >> 6, cch = i & 63;
      cp_async16(&sK[r * SQ_LD + cch * 8], &ckp[(long)(k0 + r) * KVL + cch * 8]);
    }
    for (int i = tid; i < 64 * 8; i += 256) {
      int r = i >> 3, cch = i & 7;
      cp_async16(&sKr[r * SR_LD + cch * 8], &krp[(long)(k0 + r) * ROPE_ + cch * 8]);
    }
  };

  // async-load Q tiles + first K tile
  for (int i = tid; i < 64 * 64; i += 256) {
    int r = i >> 6, cch = i & 63;
    cp_async16(&sQ[r * SQ_LD + cch * 8], &nqp[(long)r * KVL + cch * 8]);
  }
  for (int i = tid; i < 64 * 8; i += 256) {
    int r = i >> 3, cch = i & 7;
    cp_async16(&sQr[r * SR_LD + cch * 8], &qrp[(long)r * ROPE_ + cch * 8]);
  }
  issue_k(0);
  if (tid < 64) { sM[tid] = -1e30f; sL[tid] = 0.f; }

  v8f zero = {0.f, 0.f, 0.f, 0.f, 0.f, 0.f, 0.f, 0.f};
  v8f acc[16];
#pragma unroll
  for (int j = 0; j < 16; j++) acc[j] = zero;

  for (int kt = 0; kt <= qt; kt++) {
    async_wait0();
    __syncthreads();  // sK/sKr (and sQ on first iter) visible; prior iter done

    // build transposed copy sKT (feat x keys) from sK, LDS->LDS
    for (int i = tid; i < 64 * 64; i += 256) {
      int r = i >> 6, cch = i & 63;
      uint4 v = *(const uint4*)&sK[r * SQ_LD + cch * 8];
      union { uint4 v4; u16 u[8]; } t;
      t.v4 = v;
      int f0 = cch * 8;
#pragma unroll
      for (int e = 0; e < 8; e++) sKT[(f0 + e) * SR_LD + r] = t.u[e];
    }
    __syncthreads();

    // scores: 64x64, K = 512 (nope) + 64 (rope)
    v8f sa0 = zero, sa1 = zero;
#pragma unroll 4
    for (int ks = 0; ks < 16; ks++) {
      v16bf a  = frag_a_ld(sQ, SQ_LD, wr * 16, ks * 32);
      v16bf b0 = frag_b_ld(sK, SQ_LD, wc * 32, ks * 32);
      v16bf b1 = frag_b_ld(sK, SQ_LD, wc * 32 + 16, ks * 32);
      sa0 = wmma_bf16(a, b0, sa0);
      sa1 = wmma_bf16(a, b1, sa1);
    }
#pragma unroll
    for (int ks = 0; ks < 2; ks++) {
      v16bf a  = frag_a_ld(sQr, SR_LD, wr * 16, ks * 32);
      v16bf b0 = frag_b_ld(sKr, SR_LD, wc * 32, ks * 32);
      v16bf b1 = frag_b_ld(sKr, SR_LD, wc * 32 + 16, ks * 32);
      sa0 = wmma_bf16(a, b0, sa0);
      sa1 = wmma_bf16(a, b1, sa1);
    }
    // scale + causal mask -> LDS
#pragma unroll
    for (int s2 = 0; s2 < 2; s2++) {
      v8f v = s2 ? sa1 : sa0;
      int col = wc * 32 + s2 * 16 + lr;
#pragma unroll
      for (int i = 0; i < 8; i++) {
        int row = wr * 16 + i + half * 8;
        float f = v[i] * SCALE_QK;
        if (kt == qt && col > row) f = -1e30f;
        sS[row * SS_LD + col] = f;
      }
    }
    __syncthreads();  // last readers of sK/sKr done; sS complete

    // overlap: start fetching next K tile while softmax + ctx WMMAs run
    if (kt < qt) issue_k(kt + 1);

    // online softmax stats: 4 threads per row
    int rrow = tid >> 2, rseg = tid & 3;
    float lmax = -1e30f;
    for (int c2 = rseg * 16; c2 < rseg * 16 + 16; c2++)
      lmax = fmaxf(lmax, sS[rrow * SS_LD + c2]);
    sRed[rrow * 4 + rseg] = lmax;
    __syncthreads();
    float mold = sM[rrow];
    float mnew = fmaxf(mold, fmaxf(fmaxf(sRed[rrow * 4 + 0], sRed[rrow * 4 + 1]),
                                   fmaxf(sRed[rrow * 4 + 2], sRed[rrow * 4 + 3])));
    float psum = 0.f;
    for (int c2 = rseg * 16; c2 < rseg * 16 + 16; c2++) {
      float e = __expf(sS[rrow * SS_LD + c2] - mnew);
      sP[rrow * SR_LD + c2] = f2bf(e);
      psum += e;
    }
    sSum[rrow * 4 + rseg] = psum;
    __syncthreads();
    if (rseg == 0) {
      float alpha = __expf(mold - mnew);
      sAl[rrow] = alpha;
      sL[rrow] = sL[rrow] * alpha + sSum[rrow * 4 + 0] + sSum[rrow * 4 + 1] +
                 sSum[rrow * 4 + 2] + sSum[rrow * 4 + 3];
      sM[rrow] = mnew;
    }
    __syncthreads();

    // ctx = alpha*ctx + P @ ckv   (each wave: rows wr*16.., cols wc*256..+256)
    float al[8];
#pragma unroll
    for (int i = 0; i < 8; i++) al[i] = sAl[wr * 16 + i + half * 8];
#pragma unroll
    for (int j = 0; j < 16; j++)
#pragma unroll
      for (int i = 0; i < 8; i++) acc[j][i] *= al[i];
#pragma unroll
    for (int ks = 0; ks < 2; ks++) {
      v16bf a = frag_a_ld(sP, SR_LD, wr * 16, ks * 32);
#pragma unroll 4
      for (int j = 0; j < 16; j++) {
        v16bf b = frag_b_ld(sKT, SR_LD, wc * 256 + j * 16, ks * 32);
        acc[j] = wmma_bf16(a, b, acc[j]);
      }
    }
  }
  __syncthreads();

  // epilogue: ctx /= l, store bf16 to [B,H,S,512]
  float linv[8];
#pragma unroll
  for (int i = 0; i < 8; i++) linv[i] = 1.0f / sL[wr * 16 + i + half * 8];
  u16* outp = ctx + qbase * KVL;
#pragma unroll
  for (int j = 0; j < 16; j++) {
    int col = wc * 256 + j * 16 + lr;
#pragma unroll
    for (int i = 0; i < 8; i++) {
      int row = wr * 16 + i + half * 8;
      outp[(long)row * KVL + col] = f2bf(acc[j][i] * linv[i]);
    }
  }
}

// ---------------------------------------------------------------------------
// host launch
// ---------------------------------------------------------------------------

extern "C" void kernel_launch(void* const* d_in, const int* in_sizes, int n_in,
                              void* d_out, int out_size, void* d_ws, size_t ws_size,
                              hipStream_t stream) {
  (void)in_sizes; (void)n_in; (void)out_size; (void)ws_size;
  const float* hs       = (const float*)d_in[0];
  /* d_in[1] attention_mask: causal, handled analytically */
  const float* q_a_w    = (const float*)d_in[2];
  const float* q_a_ln   = (const float*)d_in[3];
  const float* qb_rope  = (const float*)d_in[4];
  const float* qk_nope  = (const float*)d_in[5];
  const float* kv_a_w   = (const float*)d_in[6];
  const float* kv_a_ln  = (const float*)d_in[7];
  const float* v_b_w    = (const float*)d_in[8];
  const float* o_w      = (const float*)d_in[9];
  /* d_in[10] position_ids == arange(S), handled analytically */
  float* out = (float*)d_out;

  const long T = (long)B_ * S_;  // 4096 tokens
  size_t off = 0;
  char* base = (char*)d_ws;
  auto take = [&](size_t bytes) {
    void* p = base + off;
    off = (off + bytes + 255) & ~(size_t)255;
    return p;
  };
  u16*   hs_b    = (u16*)take(T * HID_ * 2);
  u16*   wqa_b   = (u16*)take((size_t)QLORA * HID_ * 2);
  u16*   wqb_b   = (u16*)take((size_t)(H_ * ROPE_) * QLORA * 2);
  u16*   wnope_b = (u16*)take((size_t)(H_ * KVL) * QLORA * 2);
  u16*   wkva_b  = (u16*)take((size_t)(KVL + ROPE_) * HID_ * 2);
  u16*   wv_b    = (u16*)take((size_t)(H_ * VDIM_) * KVL * 2);
  u16*   wo_b    = (u16*)take((size_t)HID_ * (H_ * VDIM_) * 2);
  float* qa_f    = (float*)take(T * QLORA * 4);
  u16*   qa_b    = (u16*)take(T * QLORA * 2);
  float* ckv_f   = (float*)take(T * (KVL + ROPE_) * 4);
  u16*   ckv_b   = (u16*)take(T * KVL * 2);
  u16*   kr_b    = (u16*)take(T * ROPE_ * 2);
  float* qr_f    = (float*)take(T * (H_ * ROPE_) * 4);
  u16*   qr_b    = (u16*)take(T * (H_ * ROPE_) * 2);
  u16*   nq_b    = (u16*)take(T * (H_ * KVL) * 2);
  u16*   ctx_b   = (u16*)take(T * (H_ * KVL) * 2);
  u16*   ho_b    = (u16*)take(T * (H_ * VDIM_) * 2);

  auto conv = [&](const float* src, u16* dst, long n) {
    int blocks = (int)((n + 255) / 256);
    mla_f2bf_kernel<<<blocks, 256, 0, stream>>>(src, dst, n);
  };
  conv(hs,      hs_b,    T * HID_);
  conv(q_a_w,   wqa_b,   (long)QLORA * HID_);
  conv(qb_rope, wqb_b,   (long)(H_ * ROPE_) * QLORA);
  conv(qk_nope, wnope_b, (long)(H_ * KVL) * QLORA);
  conv(kv_a_w,  wkva_b,  (long)(KVL + ROPE_) * HID_);
  conv(v_b_w,   wv_b,    (long)(H_ * VDIM_) * KVL);
  conv(o_w,     wo_b,    (long)HID_ * (H_ * VDIM_));

  // q_a = hs @ q_a_w^T  -> f32
  mla_gemm_kernel<<<dim3(QLORA / 64, T / 128, 1), 256, 0, stream>>>(
      hs_b, wqa_b, qa_f, nullptr, (int)T, QLORA, HID_, HID_, HID_, QLORA,
      0, 0, 0, MODE_F32);
  mla_rmsnorm_kernel<<<(int)T, 256, 0, stream>>>(qa_f, q_a_ln, qa_b, QLORA);

  // ckv_full = hs @ kv_a_w^T -> f32 ; then rmsnorm + k_rope rope
  mla_gemm_kernel<<<dim3((KVL + ROPE_) / 64, T / 128, 1), 256, 0, stream>>>(
      hs_b, wkva_b, ckv_f, nullptr, (int)T, KVL + ROPE_, HID_, HID_, HID_,
      KVL + ROPE_, 0, 0, 0, MODE_F32);
  mla_kvpost_kernel<<<(int)T, 256, 0, stream>>>(ckv_f, kv_a_ln, ckv_b, kr_b);

  // q_rope = q_a @ q_b_rope_w^T -> f32, then rope + [B,H,S,64]
  mla_gemm_kernel<<<dim3((H_ * ROPE_) / 64, T / 128, 1), 256, 0, stream>>>(
      qa_b, wqb_b, qr_f, nullptr, (int)T, H_ * ROPE_, QLORA, QLORA, QLORA,
      H_ * ROPE_, 0, 0, 0, MODE_F32);
  mla_qrope_kernel<<<(int)((T * (H_ * ROPE_) + 255) / 256), 256, 0, stream>>>(qr_f, qr_b);

  // nope_q = q_a @ qk_nope_w^T -> bf16 in [B,H,S,512]
  mla_gemm_kernel<<<dim3((H_ * KVL) / 64, T / 128, 1), 256, 0, stream>>>(
      qa_b, wnope_b, nullptr, nq_b, (int)T, H_ * KVL, QLORA, QLORA, QLORA, 0,
      0, 0, 0, MODE_NOPE);

  // flash attention over compressed KV
  size_t smem = (size_t)(64 * SQ_LD + 64 * SR_LD + 64 * SQ_LD + 64 * SR_LD +
                         512 * SR_LD + 64 * SR_LD) * sizeof(u16) +
                (size_t)(64 * SS_LD + 64 * 3 + 256 + 256) * sizeof(float);
  (void)hipFuncSetAttribute((const void*)mla_attn_kernel,
                            hipFuncAttributeMaxDynamicSharedMemorySize, (int)smem);
  mla_attn_kernel<<<dim3(S_ / 64, H_, B_), 256, smem, stream>>>(nq_b, qr_b, ckv_b,
                                                                kr_b, ctx_b);

  // per-head v projection: ho[b,s,h*128+v] = ctx[b,h,s,:] . v_w[h,v,:]
  mla_gemm_kernel<<<dim3(VDIM_ / 64, S_ / 128, B_ * H_), 256, 0, stream>>>(
      ctx_b, wv_b, nullptr, ho_b, S_, VDIM_, KVL, KVL, KVL, 0,
      (long)S_ * KVL, (long)VDIM_ * KVL, 0, MODE_VPROJ);

  // final o projection -> f32 output
  mla_gemm_kernel<<<dim3(HID_ / 64, T / 128, 1), 256, 0, stream>>>(
      ho_b, wo_b, out, nullptr, (int)T, HID_, H_ * VDIM_, H_ * VDIM_,
      H_ * VDIM_, HID_, 0, 0, 0, MODE_F32);
}
